// Repro_10256381903436
// MI455X (gfx1250) — matrix operand order repaired and checked by reading, not compiled, not run on hardware
//
#include <hip/hip_runtime.h>
#include <stdint.h>

#define HIDDEN   2048
#define VOCAB    32768
#define SEQ      4608
#define CAPV     30.0f
#define IGNORE_L (-100)

#define M_TILE   64
#define SLICES   8
#define N_SLICE  (VOCAB / SLICES)    // 4096 cols per WG
#define N_PAIRS  (N_SLICE / 32)      // 128 pairs of 16-wide tiles
#define LDS_STRIDE (HIDDEN + 8)      // bf16 elems; +8 pad -> conflict-free ds_load_b128

typedef __bf16 bf16;
typedef __bf16 v16bf __attribute__((ext_vector_type(16)));
typedef float  v8f   __attribute__((ext_vector_type(8)));

union BFrag { uint4 q[2]; v16bf v; };

__device__ __forceinline__ float fast_tanh(float x) {
#if __has_builtin(__builtin_amdgcn_tanhf)
    return __builtin_amdgcn_tanhf(x);
#elif __has_builtin(__builtin_amdgcn_tanh_f32)
    return __builtin_amdgcn_tanh_f32(x);
#else
    float e = __expf(2.0f * x);
    return 1.0f - 2.0f / (e + 1.0f);
#endif
}

__global__ void zero_ws(float* ws, int n) {
    int i = blockIdx.x * blockDim.x + threadIdx.x;
    if (i < n) ws[i] = 0.0f;
}

__global__ __launch_bounds__(256) void ce_main(
    const bf16* __restrict__ weight, const float* __restrict__ hidden,
    const int* __restrict__ ids, const int* __restrict__ am,
    float* __restrict__ row_sumexp, float* __restrict__ row_lab)
{
    extern __shared__ bf16 sA[];   // [M_TILE][LDS_STRIDE] bf16
    const int tid      = threadIdx.x;
    const int row_base = blockIdx.x * M_TILE;
    const int n_base   = blockIdx.y * N_SLICE;

    // ---- stage A tile (64 x 2048 fp32 -> bf16) into LDS, once per WG ----
    {
        const int rowLocal = tid >> 2;              // 4 threads per row
        const int colBase  = (tid & 3) * (HIDDEN / 4);
        const float* src = hidden + (size_t)(row_base + rowLocal) * HIDDEN + colBase;
        bf16* dst = sA + rowLocal * LDS_STRIDE + colBase;
        #pragma unroll 4
        for (int i = 0; i < HIDDEN / 4; i += 8) {
            float4 f0 = *(const float4*)(src + i);
            float4 f1 = *(const float4*)(src + i + 4);
            union { bf16 h[8]; uint4 q; } pk;
            pk.h[0] = (bf16)f0.x; pk.h[1] = (bf16)f0.y;
            pk.h[2] = (bf16)f0.z; pk.h[3] = (bf16)f0.w;
            pk.h[4] = (bf16)f1.x; pk.h[5] = (bf16)f1.y;
            pk.h[6] = (bf16)f1.z; pk.h[7] = (bf16)f1.w;
            *(uint4*)(dst + i) = pk.q;
        }
    }
    __syncthreads();

    const int wave   = tid >> 5;
    const int lane   = tid & 31;
    const int half   = lane >> 4;       // 0/1: K-half (A/B), M+8 (C)
    const int l15    = lane & 15;
    const int mg     = (wave & 1) * 32; // 2 m-groups of 32 rows each
    const int nphase = wave >> 1;       // 4 wave groups interleave tile pairs

    // labels for the 16 rows this lane's accumulators correspond to
    // r<8 -> m-subtile mg, r>=8 -> m-subtile mg+16
    int labv[16];
    #pragma unroll
    for (int r = 0; r < 16; ++r) {
        int m = row_base + mg + (r & 7) + 8 * half + (r >= 8 ? 16 : 0);
        labv[r] = (m + 1 < SEQ && am[m + 1] != 0) ? ids[m + 1] : IGNORE_L;
    }

    float sexp[16], labneg[16];
    #pragma unroll
    for (int r = 0; r < 16; ++r) { sexp[r] = 0.0f; labneg[r] = 0.0f; }

    // A-frag bases: lane holds row; K chunks at k0+8*half and k0+16+8*half
    const bf16* aBase0 = sA + (mg + l15) * LDS_STRIDE + 8 * half;
    const bf16* aBase1 = aBase0 + 16 * LDS_STRIDE;

    // Each wave iteration: 32x32 macro-tile (2 m-subtiles x 2 n-tiles),
    // 2 A-frags + 2 B-frags feed 4 WMMAs. Double-buffered over K.
    for (int p = nphase; p < N_PAIRS; p += 4) {
        const int n0 = n_base + p * 32;
        const bf16* wRow0 = weight + (size_t)(n0 + l15) * HIDDEN + 8 * half;
        const bf16* wRow1 = wRow0 + (size_t)16 * HIDDEN;
        __builtin_prefetch(wRow0 + (size_t)128 * HIDDEN, 0, 3);  // next pair
        __builtin_prefetch(wRow1 + (size_t)128 * HIDDEN, 0, 3);

        v8f acc00 = {}, acc01 = {}, acc10 = {}, acc11 = {};
        BFrag a0[2], a1[2], b0[2], b1[2];

        // prologue: fill buffer 0
        a0[0].q[0] = *(const uint4*)(aBase0);
        a0[0].q[1] = *(const uint4*)(aBase0 + 16);
        a1[0].q[0] = *(const uint4*)(aBase1);
        a1[0].q[1] = *(const uint4*)(aBase1 + 16);
        b0[0].q[0] = *(const uint4*)(wRow0);
        b0[0].q[1] = *(const uint4*)(wRow0 + 16);
        b1[0].q[0] = *(const uint4*)(wRow1);
        b1[0].q[1] = *(const uint4*)(wRow1 + 16);

        #pragma unroll 2
        for (int k0 = 0; k0 < HIDDEN; k0 += 32) {
            const int cur = (k0 >> 5) & 1;
            const int nxt = cur ^ 1;
            const int kn  = k0 + 32;
            if (kn < HIDDEN) {   // issue next-iteration loads before consuming cur
                a0[nxt].q[0] = *(const uint4*)(aBase0 + kn);
                a0[nxt].q[1] = *(const uint4*)(aBase0 + kn + 16);
                a1[nxt].q[0] = *(const uint4*)(aBase1 + kn);
                a1[nxt].q[1] = *(const uint4*)(aBase1 + kn + 16);
                b0[nxt].q[0] = *(const uint4*)(wRow0 + kn);
                b0[nxt].q[1] = *(const uint4*)(wRow0 + kn + 16);
                b1[nxt].q[0] = *(const uint4*)(wRow1 + kn);
                b1[nxt].q[1] = *(const uint4*)(wRow1 + kn + 16);
            }
            acc00 = __builtin_amdgcn_wmma_f32_16x16x32_bf16(
                false, a0[cur].v, false, b0[cur].v, (short)0, acc00, false, false);
            acc01 = __builtin_amdgcn_wmma_f32_16x16x32_bf16(
                false, a0[cur].v, false, b1[cur].v, (short)0, acc01, false, false);
            acc10 = __builtin_amdgcn_wmma_f32_16x16x32_bf16(
                false, a1[cur].v, false, b0[cur].v, (short)0, acc10, false, false);
            acc11 = __builtin_amdgcn_wmma_f32_16x16x32_bf16(
                false, a1[cur].v, false, b1[cur].v, (short)0, acc11, false, false);
        }

        const int ncol0 = n0 + l15;
        const int ncol1 = n0 + 16 + l15;
        #pragma unroll
        for (int r = 0; r < 8; ++r) {
            float t00 = fast_tanh(acc00[r] * (1.0f / CAPV));
            float t01 = fast_tanh(acc01[r] * (1.0f / CAPV));
            float t10 = fast_tanh(acc10[r] * (1.0f / CAPV));
            float t11 = fast_tanh(acc11[r] * (1.0f / CAPV));
            sexp[r]     += __expf(CAPV * t00) + __expf(CAPV * t01);
            sexp[r + 8] += __expf(CAPV * t10) + __expf(CAPV * t11);
            if (labv[r] == ncol0)     labneg[r]     -= CAPV * t00;
            if (labv[r] == ncol1)     labneg[r]     -= CAPV * t01;
            if (labv[r + 8] == ncol0) labneg[r + 8] -= CAPV * t10;
            if (labv[r + 8] == ncol1) labneg[r + 8] -= CAPV * t11;
        }
    }

    // reduce across the 16 lanes of each half (offsets 1..8 stay within halves),
    // then accumulate per-row partials globally
    #pragma unroll
    for (int r = 0; r < 16; ++r) {
        float s = sexp[r], g = labneg[r];
        #pragma unroll
        for (int off = 1; off < 16; off <<= 1) {
            s += __shfl_xor(s, off, 32);
            g += __shfl_xor(g, off, 32);
        }
        if (l15 == 0) {
            int m = row_base + mg + (r & 7) + 8 * half + (r >= 8 ? 16 : 0);
            atomicAdd(&row_sumexp[m], s);
            atomicAdd(&row_lab[m], g);
        }
    }
}

__global__ __launch_bounds__(256) void ce_final(
    const float* __restrict__ row_sumexp, const float* __restrict__ row_lab,
    const int* __restrict__ ids, const int* __restrict__ am, float* __restrict__ out)
{
    __shared__ float red[256];
    float acc = 0.0f;
    for (int m = threadIdx.x; m < SEQ; m += 256) {
        int lab = (m + 1 < SEQ && am[m + 1] != 0) ? ids[m + 1] : IGNORE_L;
        if (lab != IGNORE_L) acc += __logf(row_sumexp[m]) + row_lab[m];
    }
    red[threadIdx.x] = acc;
    __syncthreads();
    for (int s = 128; s > 0; s >>= 1) {
        if (threadIdx.x < s) red[threadIdx.x] += red[threadIdx.x + s];
        __syncthreads();
    }
    if (threadIdx.x == 0) out[0] = red[0];
}

extern "C" void kernel_launch(void* const* d_in, const int* in_sizes, int n_in,
                              void* d_out, int out_size, void* d_ws, size_t ws_size,
                              hipStream_t stream) {
    const bf16*  weight = (const bf16*) d_in[0];   // [VOCAB, HIDDEN] bf16
    const float* hidden = (const float*)d_in[1];   // [1, SEQ, HIDDEN] f32
    const int*   ids    = (const int*)  d_in[2];   // [1, SEQ] i32
    const int*   am     = (const int*)  d_in[3];   // [1, SEQ] i32
    float* out = (float*)d_out;

    float* row_sumexp = (float*)d_ws;              // [SEQ]
    float* row_lab    = row_sumexp + SEQ;          // [SEQ]

    const int nz = 2 * SEQ;
    zero_ws<<<(nz + 255) / 256, 256, 0, stream>>>(row_sumexp, nz);

    dim3 grid(SEQ / M_TILE, SLICES);               // 72 x 8
    size_t ldsBytes = (size_t)M_TILE * LDS_STRIDE * sizeof(bf16);  // ~257 KB
    ce_main<<<grid, 256, ldsBytes, stream>>>(weight, hidden, ids, am,
                                             row_sumexp, row_lab);

    ce_final<<<1, 256, 0, stream>>>(row_sumexp, row_lab, ids, am, out);
}
